// MMoE_2113123909707
// MI455X (gfx1250) — matrix operand order repaired and checked by reading, not compiled, not run on hardware
//
#include <hip/hip_runtime.h>

#define B_SZ   16384
#define E_DIMC 512
#define D_IN   1024
#define D_HID  2048
#define N_EXP  16
#define EXP_SZ 512
#define APAD   1032   // padded LDS row stride (halves) for feat tile
#define EPAD   528    // padded LDS row stride (halves) for e1 tile

typedef __attribute__((ext_vector_type(16))) _Float16 v16h;
typedef __attribute__((ext_vector_type(8)))  float    v8f;

// ---------------------------------------------------------------------------
// CDNA5 async copy: global -> LDS through the async engine (ASYNCcnt).
// Low 32 bits of a flat LDS pointer are the DS byte address (ISA 10.2).
// ---------------------------------------------------------------------------
__device__ __forceinline__ void async_copy_b128(void* lds_dst, const void* gsrc) {
    unsigned dst = (unsigned)(size_t)lds_dst;
    asm volatile("global_load_async_to_lds_b128 %0, %1, off"
                 :: "v"(dst), "v"(gsrc) : "memory");
}
__device__ __forceinline__ void wait_asynccnt0() {
    asm volatile("s_wait_asynccnt 0x0" ::: "memory");
}

// ---------------------------------------------------------------------------
// WMMA fragment layouts (CDNA5 ISA 7.12.2, wave32):
// A 16x32 f16: lane = g*16+r; halves 0..7 -> K=8g..8g+7, 8..15 -> K=16+8g..16+8g+7
// B 32x16 f16: lane = g*16+n; halves 0..7 -> K=8g..8g+7, 8..15 -> K=16+8g..16+8g+7
// C/D 16x16 f32: vgpr j, lane g*16+n -> M = j+8g, N = n
// Weights pre-packed so one lane's B fragment is 32 contiguous bytes.
// ---------------------------------------------------------------------------
__device__ __forceinline__ v16h load_a_f16(const _Float16* Arow, int lda, int k0, int lane) {
    int g = lane >> 4, r = lane & 15;
    const _Float16* base = Arow + (size_t)r * lda + k0 + 8 * g;
    v16h out;
#pragma unroll
    for (int i = 0; i < 8; ++i) out[i] = base[i];
#pragma unroll
    for (int i = 0; i < 8; ++i) out[8 + i] = base[16 + i];
    return out;
}

__device__ __forceinline__ v8f wmma_f16(v16h a, v16h b, v8f c) {
    return __builtin_amdgcn_wmma_f32_16x16x32_f16(false, a, false, b, (short)0, c, false, false);
}

// ---------------------------------------------------------------------------
// Pre-pack weights: f32 [T,K,N] -> f16 fragment-ordered [T][N/16][K/32][512]
// grid = (K/32, N/16, T), 512 threads = one tile; no integer div/mod.
// ---------------------------------------------------------------------------
__global__ void pack_w_kernel(const float* __restrict__ W, _Float16* __restrict__ Wp,
                              int K, int N) {
    const int ktl = blockIdx.x, ntl = blockIdx.y, Ti = blockIdx.z;
    const int slot = threadIdx.x;                   // 0..511
    const int lane = slot >> 4, h = slot & 15, g = lane >> 4;
    const int n = ntl * 16 + (lane & 15);
    const int k = ktl * 32 + (h < 8 ? 8 * g + h : 16 + 8 * g + (h - 8));
    const int nkt = K >> 5;
    const size_t dst = (((size_t)Ti * gridDim.y + ntl) * nkt + ktl) * 512 + slot;
    Wp[dst] = (_Float16)W[(size_t)Ti * K * N + (size_t)k * N + n];
}

// ---------------------------------------------------------------------------
// pack concat(emb_paper, emb_reviewer) -> f16 [B, 1024]
// ---------------------------------------------------------------------------
__global__ void pack_x_kernel(const float* __restrict__ p, const float* __restrict__ r,
                              _Float16* __restrict__ A16) {
    int i = blockIdx.x * blockDim.x + threadIdx.x;
    if (i >= B_SZ * E_DIMC) return;
    int b = i >> 9, j = i & (E_DIMC - 1);
    A16[(size_t)b * D_IN + j]          = (_Float16)p[i];
    A16[(size_t)b * D_IN + E_DIMC + j] = (_Float16)r[i];
}

// ---------------------------------------------------------------------------
// Generic WMMA GEMM, block tile 128x128, 256 thr / 8 waves.
// Wave w owns n-tile (bx*8+w) -> B read once per block, direct from packed f16.
// A staged per k-step into LDS (fragment order) with ASYNC copies, ping-pong;
// async prefetch of next buffer overlaps the WMMA stream.
// ---------------------------------------------------------------------------
__device__ __forceinline__ void stage_a_async(_Float16* dst, const _Float16* A, int lda,
                                              int m0, int k0, int tid) {
    int mtile = tid >> 5, lane = tid & 31;
    int g = lane >> 4, r = lane & 15;
    const _Float16* src = A + (size_t)(m0 + mtile * 16 + r) * lda + k0 + 8 * g;
    _Float16* d = dst + tid * 16;
    async_copy_b128(d, src);
    async_copy_b128(d + 8, src + 16);
}

__global__ void gemm_wmma_kernel(const _Float16* __restrict__ A, int lda,
                                 const _Float16* __restrict__ Wp_base,
                                 const float* __restrict__ bbase,
                                 const int* __restrict__ task,
                                 long long wstride, long long bstride,
                                 int K, int N,
                                 float* __restrict__ out32,
                                 _Float16* __restrict__ out16,
                                 int ldc, int do_relu) {
    __shared__ __align__(32) _Float16 ash[2][4096];
    const int t = *task;
    const _Float16* Wp = Wp_base + (size_t)t * wstride;
    const float* bias  = bbase + (size_t)t * bstride;
    const int tid = threadIdx.x, lane = tid & 31, wave = tid >> 5;
    const int m0 = blockIdx.y * 128;
    const int ntile = blockIdx.x * 8 + wave;
    const int nk = K >> 5;
    const v16h* bp = (const v16h*)(Wp + (size_t)ntile * nk * 512) + lane;

    v8f acc[8];
#pragma unroll
    for (int mt = 0; mt < 8; ++mt)
#pragma unroll
        for (int j = 0; j < 8; ++j) acc[mt][j] = 0.f;

    stage_a_async(ash[0], A, lda, m0, 0, tid);
    wait_asynccnt0();
    __syncthreads();
    for (int kt = 0; kt < nk; ++kt) {
        if (kt + 1 < nk) stage_a_async(ash[(kt + 1) & 1], A, lda, m0, (kt + 1) * 32, tid);
        const v16h* af = (const v16h*)ash[kt & 1];
        v16h b = *bp; bp += 32;
#pragma unroll
        for (int mt = 0; mt < 8; ++mt)
            acc[mt] = wmma_f16(af[mt * 32 + lane], b, acc[mt]);
        wait_asynccnt0();   // my async stores to next buffer are visible
        __syncthreads();    // everyone's staging complete
    }

    const int g = lane >> 4, ln = lane & 15;
    const int n = ntile * 16 + ln;
    const float bv = bias[n];
    if (out32) {
#pragma unroll
        for (int mt = 0; mt < 8; ++mt)
#pragma unroll
            for (int j = 0; j < 8; ++j) {
                int m = m0 + mt * 16 + j + 8 * g;
                float v = acc[mt][j] + bv;
                if (do_relu) v = fmaxf(v, 0.f);
                out32[(size_t)m * ldc + n] = v;
            }
    } else {
#pragma unroll
        for (int mt = 0; mt < 8; ++mt)
#pragma unroll
            for (int j = 0; j < 8; ++j) {
                int m = m0 + mt * 16 + j + 8 * g;
                float v = acc[mt][j] + bv;
                if (do_relu) v = fmaxf(v, 0.f);
                out16[(size_t)m * ldc + n] = (_Float16)v;
            }
    }
}

// ---------------------------------------------------------------------------
// LayerNorm (+optional ReLU), f32 in -> f16 out. One block per row.
// ---------------------------------------------------------------------------
__global__ void ln_kernel(const float* __restrict__ in, const float* __restrict__ gamma,
                          const float* __restrict__ beta, _Float16* __restrict__ out,
                          int N, int do_relu) {
    const int row = blockIdx.x;
    const float* x = in + (size_t)row * N;
    float s = 0.f, ss = 0.f;
    for (int i = threadIdx.x; i < N; i += 256) { float v = x[i]; s += v; ss += v * v; }
    __shared__ float r1[256], r2[256];
    r1[threadIdx.x] = s; r2[threadIdx.x] = ss;
    __syncthreads();
    for (int off = 128; off > 0; off >>= 1) {
        if (threadIdx.x < off) { r1[threadIdx.x] += r1[threadIdx.x + off];
                                 r2[threadIdx.x] += r2[threadIdx.x + off]; }
        __syncthreads();
    }
    const float mean = r1[0] / N;
    const float var  = r2[0] / N - mean * mean;
    const float rstd = rsqrtf(var + 1e-5f);
    for (int i = threadIdx.x; i < N; i += 256) {
        float v = (x[i] - mean) * rstd * gamma[i] + beta[i];
        if (do_relu) v = fmaxf(v, 0.f);
        out[(size_t)row * N + i] = (_Float16)v;
    }
}

// ---------------------------------------------------------------------------
// Gate logits (128->16) + top-3 softmax. One thread per row.
// ---------------------------------------------------------------------------
__global__ void gate_topk_kernel(const _Float16* __restrict__ g16,
                                 const float* __restrict__ gW2, const float* __restrict__ gb2,
                                 const int* __restrict__ task,
                                 float* __restrict__ gates_ws, float* __restrict__ gates_out) {
    int row = blockIdx.x * blockDim.x + threadIdx.x;
    if (row >= B_SZ) return;
    const int t = *task;
    const float* W = gW2 + (size_t)t * 128 * N_EXP;
    const float* bb = gb2 + (size_t)t * N_EXP;
    const _Float16* gr = g16 + (size_t)row * 128;
    float logit[N_EXP];
#pragma unroll
    for (int n = 0; n < N_EXP; ++n) logit[n] = bb[n];
    for (int k = 0; k < 128; ++k) {
        float gv = (float)gr[k];
#pragma unroll
        for (int n = 0; n < N_EXP; ++n) logit[n] += gv * W[k * N_EXP + n];
    }
    int i0 = 0;
    for (int n = 1; n < N_EXP; ++n) if (logit[n] > logit[i0]) i0 = n;
    int i1 = (i0 == 0) ? 1 : 0;
    for (int n = 0; n < N_EXP; ++n) if (n != i0 && logit[n] > logit[i1]) i1 = n;
    int i2 = -1;
    for (int n = 0; n < N_EXP; ++n) if (n != i0 && n != i1 && (i2 < 0 || logit[n] > logit[i2])) i2 = n;
    float m = logit[i0];
    float e0 = __expf(logit[i0] - m), e1 = __expf(logit[i1] - m), e2 = __expf(logit[i2] - m);
    float inv = 1.f / (e0 + e1 + e2);
    float gout[N_EXP];
#pragma unroll
    for (int n = 0; n < N_EXP; ++n) gout[n] = 0.f;
    gout[i0] = e0 * inv; gout[i1] = e1 * inv; gout[i2] = e2 * inv;
#pragma unroll
    for (int n = 0; n < N_EXP; ++n) {
        gates_ws[(size_t)row * N_EXP + n]  = gout[n];
        gates_out[(size_t)row * N_EXP + n] = gout[n];
    }
}

// ---------------------------------------------------------------------------
// Fused expert stack + gated mix. 64-row tile, 512 thr (16 waves), grid B/64.
// feat tile (64x1024 f16, async-loaded) + e1 tile (64x512 f16) live in LDS
// (~200 KB of the WGP's 320 KB). Wave w owns n-range [32w, 32w+32) so each
// packed weight byte is read exactly once per block. 2 barriers per expert.
// ---------------------------------------------------------------------------
__global__ void expert_mix_kernel(const _Float16* __restrict__ feat16,
                                  const _Float16* __restrict__ We1p,
                                  const float* __restrict__ be1,
                                  const _Float16* __restrict__ We2p,
                                  const float* __restrict__ be2,
                                  const float* __restrict__ gates,
                                  _Float16* __restrict__ mix16) {
    extern __shared__ __align__(32) char smem[];
    _Float16* Afeat = (_Float16*)smem;                 // [64][APAD]
    _Float16* e1s   = Afeat + 64 * APAD;               // [64][EPAD]
    float*    gsh   = (float*)(e1s + 64 * EPAD);       // [64][16]

    const int row0 = blockIdx.x * 64;
    const int tid  = threadIdx.x;
    const int lane = tid & 31;
    const int wave = tid >> 5;                         // 0..15
    const int g = lane >> 4, ln = lane & 15;
    const int n0w = wave * 32;

    // async-stage feat tile (64 x 1024 halves), row-major with pad
    for (int idx = tid; idx < 64 * 128; idx += 512) {  // 128 x 16B per row
        int rr = idx >> 7, c = idx & 127;
        async_copy_b128(Afeat + (size_t)rr * APAD + c * 8,
                        feat16 + (size_t)(row0 + rr) * D_IN + c * 8);
    }
    for (int idx = tid; idx < 64 * N_EXP; idx += 512) {
        int rr = idx >> 4, e = idx & 15;
        gsh[idx] = gates[(size_t)(row0 + rr) * N_EXP + e];
    }
    wait_asynccnt0();
    __syncthreads();

    v8f accmix[4][2];
#pragma unroll
    for (int mt = 0; mt < 4; ++mt)
#pragma unroll
        for (int c = 0; c < 2; ++c)
#pragma unroll
            for (int j = 0; j < 8; ++j) accmix[mt][c][j] = 0.f;

    const int nk1 = D_IN >> 5;     // 32
    const int nk2 = EXP_SZ >> 5;   // 16

    for (int e = 0; e < N_EXP; ++e) {
        const float* b1e = be1 + (size_t)e * EXP_SZ;
        const float* b2e = be2 + (size_t)e * EXP_SZ;
        const v16h* w1c0 = (const v16h*)(We1p + ((size_t)e * 32 + 2 * wave)     * nk1 * 512) + lane;
        const v16h* w1c1 = (const v16h*)(We1p + ((size_t)e * 32 + 2 * wave + 1) * nk1 * 512) + lane;
        const v16h* w2c0 = (const v16h*)(We2p + ((size_t)e * 32 + 2 * wave)     * nk2 * 512) + lane;
        const v16h* w2c1 = (const v16h*)(We2p + ((size_t)e * 32 + 2 * wave + 1) * nk2 * 512) + lane;

        v8f acc[4][2];
#pragma unroll
        for (int mt = 0; mt < 4; ++mt)
#pragma unroll
            for (int c = 0; c < 2; ++c)
#pragma unroll
                for (int j = 0; j < 8; ++j) acc[mt][c][j] = 0.f;

        for (int kt = 0; kt < nk1; ++kt) {
            v16h af[4];
#pragma unroll
            for (int mt = 0; mt < 4; ++mt)
                af[mt] = load_a_f16(Afeat + (size_t)mt * 16 * APAD, APAD, kt * 32, lane);
            v16h b0 = w1c0[kt * 32];
            v16h b1 = w1c1[kt * 32];
#pragma unroll
            for (int mt = 0; mt < 4; ++mt) {
                acc[mt][0] = wmma_f16(af[mt], b0, acc[mt][0]);
                acc[mt][1] = wmma_f16(af[mt], b1, acc[mt][1]);
            }
        }
        __syncthreads();   // previous expert's GEMM2 reads of e1s complete
#pragma unroll
        for (int c = 0; c < 2; ++c) {
            int n = n0w + c * 16 + ln;
            float bv = b1e[n];
#pragma unroll
            for (int mt = 0; mt < 4; ++mt)
#pragma unroll
                for (int j = 0; j < 8; ++j) {
                    int m = mt * 16 + j + 8 * g;
                    e1s[(size_t)m * EPAD + n] = (_Float16)fmaxf(acc[mt][c][j] + bv, 0.f);
                }
        }
        __syncthreads();   // e1 tile ready

#pragma unroll
        for (int mt = 0; mt < 4; ++mt)
#pragma unroll
            for (int c = 0; c < 2; ++c)
#pragma unroll
                for (int j = 0; j < 8; ++j) acc[mt][c][j] = 0.f;

        for (int kt = 0; kt < nk2; ++kt) {
            v16h af[4];
#pragma unroll
            for (int mt = 0; mt < 4; ++mt)
                af[mt] = load_a_f16(e1s + (size_t)mt * 16 * EPAD, EPAD, kt * 32, lane);
            v16h b0 = w2c0[kt * 32];
            v16h b1 = w2c1[kt * 32];
#pragma unroll
            for (int mt = 0; mt < 4; ++mt) {
                acc[mt][0] = wmma_f16(af[mt], b0, acc[mt][0]);
                acc[mt][1] = wmma_f16(af[mt], b1, acc[mt][1]);
            }
        }
#pragma unroll
        for (int c = 0; c < 2; ++c) {
            int n = n0w + c * 16 + ln;
            float bv = b2e[n];
#pragma unroll
            for (int mt = 0; mt < 4; ++mt)
#pragma unroll
                for (int j = 0; j < 8; ++j) {
                    int m = mt * 16 + j + 8 * g;
                    accmix[mt][c][j] += gsh[m * N_EXP + e] * (acc[mt][c][j] + bv);
                }
        }
    }
#pragma unroll
    for (int c = 0; c < 2; ++c) {
        int n = n0w + c * 16 + ln;
#pragma unroll
        for (int mt = 0; mt < 4; ++mt)
#pragma unroll
            for (int j = 0; j < 8; ++j) {
                int m = mt * 16 + j + 8 * g;
                mix16[(size_t)(row0 + m) * EXP_SZ + n] = (_Float16)accmix[mt][c][j];
            }
    }
}

// ---------------------------------------------------------------------------
// Final tower layer: out = sigmoid(t2 @ tW3[t] + tb3[t]); also emit task_idx.
// ---------------------------------------------------------------------------
__global__ void tower_out_kernel(const _Float16* __restrict__ t2,
                                 const float* __restrict__ tW3, const float* __restrict__ tb3,
                                 const int* __restrict__ task, float* __restrict__ out) {
    int row = blockIdx.x * blockDim.x + threadIdx.x;
    if (row >= B_SZ) return;
    const int t = *task;
    const float* w3 = tW3 + (size_t)t * 128;
    const _Float16* xr = t2 + (size_t)row * 128;
    float s = tb3[t];
#pragma unroll 4
    for (int k = 0; k < 128; ++k) s += (float)xr[k] * w3[k];
    out[row] = 1.f / (1.f + __expf(-s));
    if (row == 0) out[B_SZ] = (float)t;
}

// ---------------------------------------------------------------------------
extern "C" void kernel_launch(void* const* d_in, const int* in_sizes, int n_in,
                              void* d_out, int out_size, void* d_ws, size_t ws_size,
                              hipStream_t stream) {
    const float* emb_p  = (const float*)d_in[0];
    const float* emb_r  = (const float*)d_in[1];
    const float* enc_W1 = (const float*)d_in[2];
    const float* enc_b1 = (const float*)d_in[3];
    const float* ln1_g  = (const float*)d_in[4];
    const float* ln1_b  = (const float*)d_in[5];
    const float* enc_W2 = (const float*)d_in[6];
    const float* enc_b2 = (const float*)d_in[7];
    const float* ln2_g  = (const float*)d_in[8];
    const float* ln2_b  = (const float*)d_in[9];
    const float* gW1    = (const float*)d_in[10];
    const float* gb1    = (const float*)d_in[11];
    const float* gW2    = (const float*)d_in[12];
    const float* gb2    = (const float*)d_in[13];
    const float* We1    = (const float*)d_in[14];
    const float* be1    = (const float*)d_in[15];
    const float* We2    = (const float*)d_in[16];
    const float* be2    = (const float*)d_in[17];
    const float* tW1    = (const float*)d_in[18];
    const float* tb1    = (const float*)d_in[19];
    const float* tW2    = (const float*)d_in[20];
    const float* tb2    = (const float*)d_in[21];
    const float* tW3    = (const float*)d_in[22];
    const float* tb3    = (const float*)d_in[23];
    const int*   task   = (const int*)d_in[24];

    float* out = (float*)d_out;
    char*  ws  = (char*)d_ws;

    // activation workspace (reuse; peak 224MB) -------------------------------
    _Float16* A16    = (_Float16*)(ws + 0);                 // [B,1024] f16
    float*    z      = (float*)(ws + (32ull << 20));        // [B,2048] f32
    _Float16* h16    = (_Float16*)(ws + (160ull << 20));    // [B,2048] f16
    float*    z2     = (float*)(ws + (32ull << 20));        // [B,1024] f32 (reuse z)
    _Float16* feat16 = (_Float16*)(ws + 0);                 // [B,1024] f16 (reuse A16)
    _Float16* g16    = (_Float16*)(ws + (32ull << 20));     // [B,128] f16
    float*    gates  = (float*)(ws + (37ull << 20));        // [B,16] f32
    _Float16* mix16  = (_Float16*)(ws + (38ull << 20));     // [B,512] f16
    _Float16* t1     = (_Float16*)(ws + (54ull << 20));     // [B,256] f16
    _Float16* t2     = (_Float16*)(ws + (62ull << 20));     // [B,128] f16
    // packed f16 weights ------------------------------------------------------
    _Float16* We1p   = (_Float16*)(ws + (224ull << 20));    // 16MB
    _Float16* We2p   = (_Float16*)(ws + (240ull << 20));    // 8MB
    _Float16* eW1p   = (_Float16*)(ws + (248ull << 20));    // 4MB
    _Float16* eW2p   = (_Float16*)(ws + (252ull << 20));    // 4MB
    _Float16* gW1p   = (_Float16*)(ws + (256ull << 20));    // 0.5MB
    _Float16* tW1p   = (_Float16*)(ws + (257ull << 20));    // 0.5MB
    _Float16* tW2p   = (_Float16*)(ws + (258ull << 20));    // 0.125MB

    // weight pre-pack (f32 -> f16, fragment order), div-free 3D grids
    pack_w_kernel<<<dim3(D_IN / 32, EXP_SZ / 16, 16), 512, 0, stream>>>(We1, We1p, D_IN, EXP_SZ);
    pack_w_kernel<<<dim3(EXP_SZ / 32, EXP_SZ / 16, 16), 512, 0, stream>>>(We2, We2p, EXP_SZ, EXP_SZ);
    pack_w_kernel<<<dim3(D_IN / 32, D_HID / 16, 1), 512, 0, stream>>>(enc_W1, eW1p, D_IN, D_HID);
    pack_w_kernel<<<dim3(D_HID / 32, D_IN / 16, 1), 512, 0, stream>>>(enc_W2, eW2p, D_HID, D_IN);
    pack_w_kernel<<<dim3(D_IN / 32, 128 / 16, 2), 512, 0, stream>>>(gW1, gW1p, D_IN, 128);
    pack_w_kernel<<<dim3(EXP_SZ / 32, 256 / 16, 2), 512, 0, stream>>>(tW1, tW1p, EXP_SZ, 256);
    pack_w_kernel<<<dim3(256 / 32, 128 / 16, 2), 512, 0, stream>>>(tW2, tW2p, 256, 128);

    pack_x_kernel<<<(B_SZ * E_DIMC) / 256, 256, 0, stream>>>(emb_p, emb_r, A16);

    // encoder
    gemm_wmma_kernel<<<dim3(D_HID / 128, B_SZ / 128), 256, 0, stream>>>(
        A16, D_IN, eW1p, enc_b1, task, 0, 0, D_IN, D_HID, z, nullptr, D_HID, 0);
    ln_kernel<<<B_SZ, 256, 0, stream>>>(z, ln1_g, ln1_b, h16, D_HID, 1);
    gemm_wmma_kernel<<<dim3(D_IN / 128, B_SZ / 128), 256, 0, stream>>>(
        h16, D_HID, eW2p, enc_b2, task, 0, 0, D_HID, D_IN, z2, nullptr, D_IN, 0);
    ln_kernel<<<B_SZ, 256, 0, stream>>>(z2, ln2_g, ln2_b, feat16, D_IN, 0);

    // gates
    gemm_wmma_kernel<<<dim3(1, B_SZ / 128), 256, 0, stream>>>(
        feat16, D_IN, gW1p, gb1, task, (long long)D_IN * 128, 128, D_IN, 128,
        nullptr, g16, 128, 1);
    gate_topk_kernel<<<B_SZ / 256, 256, 0, stream>>>(g16, gW2, gb2, task,
                                                     gates, out + B_SZ + 1);

    // fused experts + gated mix (dynamic LDS ~199KB: 320KB-LDS WGP feature)
    {
        size_t smem = (size_t)(64 * APAD + 64 * EPAD) * sizeof(_Float16)
                    + (size_t)(64 * N_EXP) * sizeof(float);
        expert_mix_kernel<<<B_SZ / 64, 512, smem, stream>>>(feat16, We1p, be1,
                                                            We2p, be2, gates, mix16);
    }

    // towers
    gemm_wmma_kernel<<<dim3(256 / 128, B_SZ / 128), 256, 0, stream>>>(
        mix16, EXP_SZ, tW1p, tb1, task, (long long)EXP_SZ * 256, 256, EXP_SZ, 256,
        nullptr, t1, 256, 1);
    gemm_wmma_kernel<<<dim3(1, B_SZ / 128), 256, 0, stream>>>(
        t1, 256, tW2p, tb2, task, (long long)256 * 128, 128, 256, 128,
        nullptr, t2, 128, 1);
    tower_out_kernel<<<B_SZ / 256, 256, 0, stream>>>(t2, tW3, tb3, task, out);
}
